// PrometheusBlock_45243185496831
// MI455X (gfx1250) — matrix operand order repaired and checked by reading, not compiled, not run on hardware
//
// PrometheusBlock for MI455X (gfx1250).
//  - Heavy token GEMMs: split-bf16 "3-product" WMMA (v_wmma_f32_16x16x32_bf16,
//    hi*hi + hi*lo + lo*hi, f32 accumulate).
//    Preferred path: operands pre-decomposed into hi/lo bf16 planes (B planes
//    pre-transposed [N][K]) so GEMM staging is a pure byte copy done with
//    GLOBAL_LOAD_ASYNC_TO_LDS_B128 + s_wait_asynccnt, double-buffered in LDS.
//    Fallback path (small workspace): fused f32->bf16 split during staging.
//  - expm scaling-and-squaring chain: exact f32 WMMA (v_wmma_f32_16x16x4_f32).
//  - Fused epilogues: exact GELU (optionally emitting hi/lo planes),
//    per-row MoE accumulate, scale+identity for the expm Horner recurrence.
#include <hip/hip_runtime.h>
#include <math.h>

typedef __attribute__((ext_vector_type(2)))  float  v2f;
typedef __attribute__((ext_vector_type(4)))  float  v4f;
typedef __attribute__((ext_vector_type(8)))  float  v8f;
typedef __attribute__((ext_vector_type(4)))  __bf16 v4bf;
typedef __attribute__((ext_vector_type(8)))  __bf16 v8bf;
typedef __attribute__((ext_vector_type(16))) __bf16 v16bf;

constexpr int NTOK   = 4096;   // B*S
constexpr int DMODEL = 1024;
constexpr int DFF    = 4096;
constexpr int NEXP   = 8;
constexpr float LN_EPS = 1e-5f;

enum { EPI_NONE = 0, EPI_GELU = 1, EPI_MOE = 2, EPI_SCALE_IDENT = 3,
       EPI_GELU_SPLIT = 4 };

__device__ __forceinline__ float gelu_exact(float v) {
  return 0.5f * v * (1.0f + erff(v * 0.70710678118654752440f));
}

__device__ __forceinline__ void split_bf16(float v, __bf16& hi, __bf16& lo) {
  __bf16 h = (__bf16)v;            // RNE to 8 mantissa bits
  hi = h;
  lo = (__bf16)(v - (float)h);     // next ~8 bits
}

// gfx1250 async copy: 16B global -> LDS per lane, tracked by ASYNCcnt.
__device__ __forceinline__ void async_copy_b128(void* lds, const void* gsrc) {
  unsigned off = (unsigned)(uintptr_t)lds;   // low 32 bits = LDS byte address
  asm volatile("global_load_async_to_lds_b128 %0, %1, off"
               :: "v"(off), "v"(gsrc) : "memory");
}
__device__ __forceinline__ void wait_async0() {
  asm volatile("s_wait_asynccnt 0x0" ::: "memory");
}

// Shared fused epilogue. C/D layout: VGPR v -> row (v + 8*(lane>=16)), col=lane%16.
template<int EPI>
__device__ __forceinline__
void gemm_epilogue(v8f acc[2][4], float* __restrict__ C,
                   __bf16* __restrict__ Ch, __bf16* __restrict__ Cl,
                   const float* __restrict__ bias,
                   const float* __restrict__ rowscale, float scale,
                   int N, int waveM, int waveN, int lr, int lhalf)
{
#pragma unroll
  for (int mt = 0; mt < 2; ++mt) {
#pragma unroll
    for (int nt = 0; nt < 4; ++nt) {
      const int gn = blockIdx.x * 128 + waveN * 64 + nt * 16 + lr;
      const float bi = (EPI == EPI_GELU || EPI == EPI_MOE ||
                        EPI == EPI_GELU_SPLIT) ? bias[gn] : 0.0f;
#pragma unroll
      for (int v = 0; v < 8; ++v) {
        const int gm = blockIdx.y * 128 + waveM * 32 + mt * 16 + lhalf * 8 + v;
        const size_t idx = (size_t)gm * N + gn;
        float val = acc[mt][nt][v] + bi;
        if (EPI == EPI_GELU) {
          C[idx] = gelu_exact(val);
        } else if (EPI == EPI_GELU_SPLIT) {
          __bf16 hh, ll; split_bf16(gelu_exact(val), hh, ll);
          Ch[idx] = hh; Cl[idx] = ll;
        } else if (EPI == EPI_MOE) {
          C[idx] += rowscale[gm] * val;     // moe += wdense[m,e]*(h@w2+b2)
        } else if (EPI == EPI_SCALE_IDENT) {
          C[idx] = val * scale + ((gm == gn) ? 1.0f : 0.0f);
        } else {
          C[idx] = val;
        }
      }
    }
  }
}

// ---------------------------------------------------------------------------
// Preferred split-bf16 GEMM: hi/lo planes in, async-LDS staged, double-buffered.
// A planes row-major [M][K]; B planes pre-transposed [N][K]. M,N%128==0, K%32==0.
// Block tile 128x128, 8 waves, wave tile 32x64, BK=32, 24 WMMAs per chunk.
// ---------------------------------------------------------------------------
template<int EPI>
__global__ __launch_bounds__(256)
void gemm_wmma_bf16x3_planes(const __bf16* __restrict__ Ahg,
                             const __bf16* __restrict__ Alg,
                             const __bf16* __restrict__ Bhg,
                             const __bf16* __restrict__ Blg,
                             float* __restrict__ C,
                             __bf16* __restrict__ Ch, __bf16* __restrict__ Cl,
                             const float* __restrict__ bias,
                             const float* __restrict__ rowscale, float scale,
                             int M, int N, int K)
{
  constexpr int BM = 128, BN = 128, BK = 32, LDT = BK + 8;
  __shared__ __bf16 Ah[2][BM][LDT], Al[2][BM][LDT];
  __shared__ __bf16 Bh[2][BN][LDT], Bl[2][BN][LDT];

  const int tid   = threadIdx.x;
  const int lane  = tid & 31;
  const int wave  = tid >> 5;
  const int waveM = wave & 3;
  const int waveN = wave >> 2;
  const int lr    = lane & 15;
  const int lhalf = lane >> 4;
  const int ka    = lhalf * 16;   // A frag: K run   {ka..ka+15}
  const int kb    = lhalf * 16;   // B frag: K run   {kb..kb+15}

  const size_t baseA = (size_t)blockIdx.y * BM * K;
  const size_t baseB = (size_t)blockIdx.x * BN * K;

  // Stage one BK chunk of all 4 planes: 4 x 128 rows x 64B, pure async copies.
  auto stage = [&](int k0, int buf) {
#pragma unroll
    for (int u = 0; u < 2; ++u) {
      const int unit = tid + u * 256;
      const int row  = unit >> 2;
      const int c    = (unit & 3) * 8;               // bf16 index, 16B units
      const size_t ga = baseA + (size_t)row * K + k0 + c;
      const size_t gb = baseB + (size_t)row * K + k0 + c;
      async_copy_b128(&Ah[buf][row][c], Ahg + ga);
      async_copy_b128(&Al[buf][row][c], Alg + ga);
      async_copy_b128(&Bh[buf][row][c], Bhg + gb);
      async_copy_b128(&Bl[buf][row][c], Blg + gb);
    }
  };

  v8f acc[2][4] = {};
  const int nch = K / BK;
  stage(0, 0);

  for (int c = 0; c < nch; ++c) {
    wait_async0();                 // chunk c resident in LDS (this wave's ops)
    __syncthreads();               // ... and every other wave's too
    if (c + 1 < nch) stage((c + 1) * BK, (c + 1) & 1);   // overlap with WMMAs
    const int buf = c & 1;

    v16bf ahf[2], alf[2], bhf[4], blf[4];
#pragma unroll
    for (int mt = 0; mt < 2; ++mt) {
      const int row = waveM * 32 + mt * 16 + lr;
      v8bf h0 = *(const v8bf*)&Ah[buf][row][ka];
      v8bf h1 = *(const v8bf*)&Ah[buf][row][ka + 8];
      v8bf l0 = *(const v8bf*)&Al[buf][row][ka];
      v8bf l1 = *(const v8bf*)&Al[buf][row][ka + 8];
      ahf[mt] = __builtin_shufflevector(h0, h1, 0,1,2,3,4,5,6,7,8,9,10,11,12,13,14,15);
      alf[mt] = __builtin_shufflevector(l0, l1, 0,1,2,3,4,5,6,7,8,9,10,11,12,13,14,15);
    }
#pragma unroll
    for (int nt = 0; nt < 4; ++nt) {
      const int col = waveN * 64 + nt * 16 + lr;
      v8bf h0 = *(const v8bf*)&Bh[buf][col][kb];
      v8bf h1 = *(const v8bf*)&Bh[buf][col][kb + 8];
      v8bf l0 = *(const v8bf*)&Bl[buf][col][kb];
      v8bf l1 = *(const v8bf*)&Bl[buf][col][kb + 8];
      bhf[nt] = __builtin_shufflevector(h0, h1, 0,1,2,3,4,5,6,7,8,9,10,11,12,13,14,15);
      blf[nt] = __builtin_shufflevector(l0, l1, 0,1,2,3,4,5,6,7,8,9,10,11,12,13,14,15);
    }
#pragma unroll
    for (int mt = 0; mt < 2; ++mt)
#pragma unroll
      for (int nt = 0; nt < 4; ++nt) {
        acc[mt][nt] = __builtin_amdgcn_wmma_f32_16x16x32_bf16(
            false, ahf[mt], false, bhf[nt], (short)0, acc[mt][nt], false, false);
        acc[mt][nt] = __builtin_amdgcn_wmma_f32_16x16x32_bf16(
            false, ahf[mt], false, blf[nt], (short)0, acc[mt][nt], false, false);
        acc[mt][nt] = __builtin_amdgcn_wmma_f32_16x16x32_bf16(
            false, alf[mt], false, bhf[nt], (short)0, acc[mt][nt], false, false);
      }
  }

  gemm_epilogue<EPI>(acc, C, Ch, Cl, bias, rowscale, scale, N,
                     waveM, waveN, lr, lhalf);
}

// ---------------------------------------------------------------------------
// Fallback split-bf16 GEMM (f32 in, conversion fused into staging).
// ---------------------------------------------------------------------------
template<int EPI>
__global__ __launch_bounds__(256)
void gemm_wmma_bf16x3(const float* __restrict__ A, const float* __restrict__ B,
                      float* __restrict__ C, const float* __restrict__ bias,
                      const float* __restrict__ rowscale, float scale,
                      int M, int N, int K)
{
  constexpr int BM = 128, BN = 128, BK = 32, LDT = BK + 8;
  __shared__ __bf16 Ah[BM][LDT], Al[BM][LDT];
  __shared__ __bf16 Bh[BN][LDT], Bl[BN][LDT];

  const int tid   = threadIdx.x;
  const int lane  = tid & 31;
  const int wave  = tid >> 5;
  const int waveM = wave & 3;
  const int waveN = wave >> 2;
  const int lr    = lane & 15;
  const int lhalf = lane >> 4;
  const int ka    = lhalf * 16;
  const int kb    = lhalf * 16;

  const int aRow = tid >> 3;
  const int aCol = (tid & 7) * 4;
  const int bRow = tid >> 5;
  const int bCol = (tid & 31) * 4;

  const float* Ablk = A + (size_t)blockIdx.y * BM * K;
  const float* Bblk = B + (size_t)blockIdx.x * BN;

  v8f acc[2][4] = {};

  for (int k0 = 0; k0 < K; k0 += BK) {
    __syncthreads();
#pragma unroll
    for (int p = 0; p < 4; ++p) {
      int r = aRow + p * 32;
      v4f av = *(const v4f*)(Ablk + (size_t)r * K + k0 + aCol);
      v4bf h, l;
#pragma unroll
      for (int j = 0; j < 4; ++j) { __bf16 hh, ll; split_bf16(av[j], hh, ll); h[j] = hh; l[j] = ll; }
      *(v4bf*)&Ah[r][aCol] = h;
      *(v4bf*)&Al[r][aCol] = l;
    }
#pragma unroll
    for (int p = 0; p < 4; ++p) {
      int r = bRow + p * 8;
      v4f bv = *(const v4f*)(Bblk + (size_t)(k0 + r) * N + bCol);
#pragma unroll
      for (int j = 0; j < 4; ++j) {
        __bf16 hh, ll; split_bf16(bv[j], hh, ll);
        Bh[bCol + j][r] = hh;
        Bl[bCol + j][r] = ll;
      }
    }
    if (k0 + BK < K) {
      __builtin_prefetch(Ablk + (size_t)aRow * K + (k0 + BK) + aCol, 0, 0);
      __builtin_prefetch(Bblk + (size_t)(k0 + BK + bRow) * N + bCol, 0, 0);
    }
    __syncthreads();

    v16bf ah[2], al[2], bh[4], bl[4];
#pragma unroll
    for (int mt = 0; mt < 2; ++mt) {
      const int row = waveM * 32 + mt * 16 + lr;
      v8bf h0 = *(const v8bf*)&Ah[row][ka];
      v8bf h1 = *(const v8bf*)&Ah[row][ka + 8];
      v8bf l0 = *(const v8bf*)&Al[row][ka];
      v8bf l1 = *(const v8bf*)&Al[row][ka + 8];
      ah[mt] = __builtin_shufflevector(h0, h1, 0,1,2,3,4,5,6,7,8,9,10,11,12,13,14,15);
      al[mt] = __builtin_shufflevector(l0, l1, 0,1,2,3,4,5,6,7,8,9,10,11,12,13,14,15);
    }
#pragma unroll
    for (int nt = 0; nt < 4; ++nt) {
      const int col = waveN * 64 + nt * 16 + lr;
      v8bf h0 = *(const v8bf*)&Bh[col][kb];
      v8bf h1 = *(const v8bf*)&Bh[col][kb + 8];
      v8bf l0 = *(const v8bf*)&Bl[col][kb];
      v8bf l1 = *(const v8bf*)&Bl[col][kb + 8];
      bh[nt] = __builtin_shufflevector(h0, h1, 0,1,2,3,4,5,6,7,8,9,10,11,12,13,14,15);
      bl[nt] = __builtin_shufflevector(l0, l1, 0,1,2,3,4,5,6,7,8,9,10,11,12,13,14,15);
    }
#pragma unroll
    for (int mt = 0; mt < 2; ++mt)
#pragma unroll
      for (int nt = 0; nt < 4; ++nt) {
        acc[mt][nt] = __builtin_amdgcn_wmma_f32_16x16x32_bf16(
            false, ah[mt], false, bh[nt], (short)0, acc[mt][nt], false, false);
        acc[mt][nt] = __builtin_amdgcn_wmma_f32_16x16x32_bf16(
            false, ah[mt], false, bl[nt], (short)0, acc[mt][nt], false, false);
        acc[mt][nt] = __builtin_amdgcn_wmma_f32_16x16x32_bf16(
            false, al[mt], false, bh[nt], (short)0, acc[mt][nt], false, false);
      }
  }

  gemm_epilogue<EPI>(acc, C, nullptr, nullptr, bias, rowscale, scale, N,
                     waveM, waveN, lr, lhalf);
}

// ---------------------------------------------------------------------------
// Exact-f32 WMMA GEMM (v_wmma_f32_16x16x4_f32) for the expm chain.
// ---------------------------------------------------------------------------
template<int EPI>
__global__ __launch_bounds__(256)
void gemm_wmma_f32(const float* __restrict__ A, const float* __restrict__ B,
                   float* __restrict__ C, const float* __restrict__ bias,
                   const float* __restrict__ rowscale, float scale,
                   int M, int N, int K)
{
  constexpr int BM = 128, BN = 128, BK = 16, PAD = 4;
  __shared__ float As[BM][BK + PAD];
  __shared__ float Bt[BN][BK + PAD];

  const int tid   = threadIdx.x;
  const int lane  = tid & 31;
  const int wave  = tid >> 5;
  const int waveM = wave & 3;
  const int waveN = wave >> 2;
  const int lr    = lane & 15;
  const int lk    = (lane >> 4) * 2;
  const int lhalf = lane >> 4;

  const int aRow = tid >> 2;
  const int aCol = (tid & 3) * 4;
  const int bRow = tid >> 5;
  const int bCol = (tid & 31) * 4;

  const float* Ablk = A + (size_t)blockIdx.y * BM * K;
  const float* Bblk = B + (size_t)blockIdx.x * BN;

  v8f acc[2][4] = {};

  for (int k0 = 0; k0 < K; k0 += BK) {
    __syncthreads();
#pragma unroll
    for (int p = 0; p < 2; ++p) {
      int r = aRow + p * 64;
      v4f av = *(const v4f*)(Ablk + (size_t)r * K + k0 + aCol);
      *(v4f*)&As[r][aCol] = av;
    }
#pragma unroll
    for (int p = 0; p < 2; ++p) {
      int r = bRow + p * 8;
      v4f bv = *(const v4f*)(Bblk + (size_t)(k0 + r) * N + bCol);
      Bt[bCol + 0][r] = bv.x;
      Bt[bCol + 1][r] = bv.y;
      Bt[bCol + 2][r] = bv.z;
      Bt[bCol + 3][r] = bv.w;
    }
    __syncthreads();

#pragma unroll
    for (int kk = 0; kk < BK; kk += 4) {
      v2f afrag[2], bfrag[4];
#pragma unroll
      for (int mt = 0; mt < 2; ++mt)
        afrag[mt] = *(const v2f*)&As[waveM * 32 + mt * 16 + lr][kk + lk];
#pragma unroll
      for (int nt = 0; nt < 4; ++nt)
        bfrag[nt] = *(const v2f*)&Bt[waveN * 64 + nt * 16 + lr][kk + lk];
#pragma unroll
      for (int mt = 0; mt < 2; ++mt)
#pragma unroll
        for (int nt = 0; nt < 4; ++nt)
          acc[mt][nt] = __builtin_amdgcn_wmma_f32_16x16x4_f32(
              false, afrag[mt], false, bfrag[nt], (short)0, acc[mt][nt],
              false, false);
    }
  }

  gemm_epilogue<EPI>(acc, C, nullptr, nullptr, bias, rowscale, scale, N,
                     waveM, waveN, lr, lhalf);
}

// ---------------------------------------------------------------------------
// Decompose f32 -> hi/lo bf16 planes (same layout).
// ---------------------------------------------------------------------------
__global__ __launch_bounds__(256)
void decompose_kernel(const float* __restrict__ X, __bf16* __restrict__ H,
                      __bf16* __restrict__ L, long n)
{
  long i = (long)blockIdx.x * 256 + threadIdx.x;
  if (i < n) {
    __bf16 h, l; split_bf16(X[i], h, l);
    H[i] = h; L[i] = l;
  }
}

// Decompose + transpose: X f32 [K][N] -> H/L bf16 [N][K]. K,N % 32 == 0.
__global__ __launch_bounds__(256)
void decompose_t_kernel(const float* __restrict__ X, __bf16* __restrict__ H,
                        __bf16* __restrict__ L, int K, int N)
{
  __shared__ float tile[32][33];
  const int n0 = blockIdx.x * 32, k0 = blockIdx.y * 32;
  const int tx = threadIdx.x & 31, ty = threadIdx.x >> 5;   // ty: 0..7
#pragma unroll
  for (int j = 0; j < 32; j += 8)
    tile[ty + j][tx] = X[(size_t)(k0 + ty + j) * N + n0 + tx];
  __syncthreads();
#pragma unroll
  for (int j = 0; j < 32; j += 8) {
    const int n = n0 + ty + j, k = k0 + tx;
    __bf16 h, l; split_bf16(tile[tx][ty + j], h, l);
    H[(size_t)n * K + k] = h;
    L[(size_t)n * K + k] = l;
  }
}

// ---------------------------------------------------------------------------
// LayerNorm / gate / top2 / elementwise / mix kernels.
// ---------------------------------------------------------------------------
__global__ __launch_bounds__(256)
void layernorm_kernel(const float* __restrict__ x, const float* __restrict__ g,
                      const float* __restrict__ b, float* __restrict__ xn)
{
  __shared__ float r1[256], r2[256];
  const int tid = threadIdx.x;
  const float* xr = x + (size_t)blockIdx.x * DMODEL;
  float s1 = 0.f, s2 = 0.f;
  for (int i = tid; i < DMODEL; i += 256) { float v = xr[i]; s1 += v; s2 += v * v; }
  r1[tid] = s1; r2[tid] = s2; __syncthreads();
  for (int off = 128; off > 0; off >>= 1) {
    if (tid < off) { r1[tid] += r1[tid + off]; r2[tid] += r2[tid + off]; }
    __syncthreads();
  }
  const float mu   = r1[0] * (1.0f / DMODEL);
  const float var  = r2[0] * (1.0f / DMODEL) - mu * mu;
  const float rinv = rsqrtf(var + LN_EPS);
  float* xo = xn + (size_t)blockIdx.x * DMODEL;
  for (int i = tid; i < DMODEL; i += 256)
    xo[i] = (xr[i] - mu) * rinv * g[i] + b[i];
}

__global__ __launch_bounds__(256)
void gate_kernel(const float* __restrict__ xn, const float* __restrict__ gw,
                 const float* __restrict__ gb, float* __restrict__ logits)
{
  const int token = blockIdx.x;
  const int lane  = threadIdx.x & 31;
  const int e     = threadIdx.x >> 5;
  const float* xr = xn + (size_t)token * DMODEL;
  float s = 0.f;
  for (int i = lane; i < DMODEL; i += 32) s += xr[i] * gw[(size_t)i * NEXP + e];
  for (int off = 16; off > 0; off >>= 1) s += __shfl_xor(s, off, 32);
  if (lane == 0) logits[(size_t)token * NEXP + e] = s + gb[e];
}

__global__ __launch_bounds__(256)
void top2_kernel(const float* __restrict__ logits, float* __restrict__ wt)
{
  const int t = blockIdx.x * blockDim.x + threadIdx.x;
  if (t >= NTOK) return;
  const float* L = logits + (size_t)t * NEXP;
  int i0 = 0; float v0 = L[0];
  for (int e = 1; e < NEXP; ++e) { float v = L[e]; if (v > v0) { v0 = v; i0 = e; } }
  int i1 = -1; float v1 = -INFINITY;
  for (int e = 0; e < NEXP; ++e) {
    if (e == i0) continue;
    float v = L[e]; if (v > v1) { v1 = v; i1 = e; }
  }
  const float e1 = expf(v1 - v0);
  const float w0 = 1.0f / (1.0f + e1);
  const float w1 = e1  / (1.0f + e1);
  for (int e = 0; e < NEXP; ++e) {
    float w = (e == i0) ? w0 : (e == i1) ? w1 : 0.0f;
    wt[(size_t)e * NTOK + t] = w;
  }
}

__global__ __launch_bounds__(256)
void scale_kernel(const float* __restrict__ X, float* __restrict__ Y,
                  float s, int n)
{
  int i = blockIdx.x * blockDim.x + threadIdx.x;
  if (i < n) Y[i] = X[i] * s;
}

__global__ __launch_bounds__(256)
void scale_ident_kernel(const float* __restrict__ X, float* __restrict__ Y,
                        float s, int d)
{
  int i = blockIdx.x * blockDim.x + threadIdx.x;
  if (i < d * d) {
    int r = i / d, c = i - r * d;
    Y[i] = X[i] * s + ((r == c) ? 1.0f : 0.0f);
  }
}

__global__ __launch_bounds__(256)
void mix_out_kernel(const float* __restrict__ x, const float* __restrict__ abel,
                    const float* __restrict__ nab, const float* __restrict__ mw,
                    const float* __restrict__ mb, float* __restrict__ out)
{
  __shared__ float red[256];
  const int tid = threadIdx.x;
  const size_t base = (size_t)blockIdx.x * DMODEL;
  const float* ar = abel + base;
  const float* nr = nab + base;
  const float* xr = x + base;
  float s = 0.f;
  for (int i = tid; i < DMODEL; i += 256)
    s += ar[i] * mw[i] + nr[i] * mw[DMODEL + i];
  red[tid] = s; __syncthreads();
  for (int off = 128; off > 0; off >>= 1) {
    if (tid < off) red[tid] += red[tid + off];
    __syncthreads();
  }
  const float g = 1.0f / (1.0f + expf(-(red[0] + mb[0])));
  for (int i = tid; i < DMODEL; i += 256)
    out[base + i] = xr[i] + ar[i] * (1.0f - g) + nr[i] * g;
}

// ---------------------------------------------------------------------------
// Host-side launchers.
// ---------------------------------------------------------------------------
template<int EPI>
static void launch_gemm_f32(const float* A, const float* B, float* C,
                            const float* bias, const float* rowscale, float scale,
                            int M, int N, int K, hipStream_t stream)
{
  dim3 grid(N / 128, M / 128), block(256);
  gemm_wmma_f32<EPI><<<grid, block, 0, stream>>>(A, B, C, bias, rowscale,
                                                 scale, M, N, K);
}

template<int EPI>
static void launch_gemm_bf3(const float* A, const float* B, float* C,
                            const float* bias, const float* rowscale, float scale,
                            int M, int N, int K, hipStream_t stream)
{
  dim3 grid(N / 128, M / 128), block(256);
  gemm_wmma_bf16x3<EPI><<<grid, block, 0, stream>>>(A, B, C, bias, rowscale,
                                                    scale, M, N, K);
}

template<int EPI>
static void launch_gemm_planes(const __bf16* Ah, const __bf16* Al,
                               const __bf16* Bh, const __bf16* Bl,
                               float* C, __bf16* Ch, __bf16* Cl,
                               const float* bias, const float* rowscale,
                               float scale, int M, int N, int K,
                               hipStream_t stream)
{
  dim3 grid(N / 128, M / 128), block(256);
  gemm_wmma_bf16x3_planes<EPI><<<grid, block, 0, stream>>>(
      Ah, Al, Bh, Bl, C, Ch, Cl, bias, rowscale, scale, M, N, K);
}

static void launch_decompose(const float* X, __bf16* H, __bf16* L, long n,
                             hipStream_t stream)
{
  decompose_kernel<<<(unsigned)((n + 255) / 256), 256, 0, stream>>>(X, H, L, n);
}

static void launch_decompose_t(const float* X, __bf16* H, __bf16* L,
                               int K, int N, hipStream_t stream)
{
  dim3 grid(N / 32, K / 32), block(256);
  decompose_t_kernel<<<grid, block, 0, stream>>>(X, H, L, K, N);
}

extern "C" void kernel_launch(void* const* d_in, const int* in_sizes, int n_in,
                              void* d_out, int out_size, void* d_ws, size_t ws_size,
                              hipStream_t stream)
{
  (void)in_sizes; (void)n_in; (void)out_size;
  const float* x       = (const float*)d_in[0];
  const float* ln_g    = (const float*)d_in[1];
  const float* ln_b    = (const float*)d_in[2];
  const float* abel_w  = (const float*)d_in[3];
  const float* abel_b  = (const float*)d_in[4];
  const float* gate_w  = (const float*)d_in[5];
  const float* gate_b  = (const float*)d_in[6];
  const float* w1      = (const float*)d_in[7];
  const float* b1      = (const float*)d_in[8];
  const float* w2      = (const float*)d_in[9];
  const float* b2      = (const float*)d_in[10];
  const float* gens    = (const float*)d_in[11];
  const float* mix_w   = (const float*)d_in[12];
  const float* mix_b   = (const float*)d_in[13];
  float* out = (float*)d_out;

  const size_t ND  = (size_t)NTOK * DMODEL;
  const size_t NDF = (size_t)NTOK * DFF;
  const size_t DD  = (size_t)DMODEL * DMODEL;
  const size_t WE  = (size_t)DMODEL * DFF;     // one expert weight matrix

  // ---- f32 region (always used) ----
  float* p = (float*)d_ws;
  float* xn      = p; p += ND;
  float* abel    = p; p += ND;
  float* moe     = p; p += ND;
  float* nonabel = p; p += ND;
  float* h_f32   = p; p += NDF;                // fallback path only
  float* logits  = p; p += (size_t)NTOK * NEXP;
  float* wdense  = p; p += (size_t)NEXP * NTOK;
  float* Xb      = p; p += DD;
  float* Pb      = p; p += DD;
  float* Tb      = p; p += DD;
  float* Em[3]   = { p, p + DD, p + 2 * DD }; p += 3 * DD;
  float* R1      = p; p += DD;
  float* R2      = p; p += DD;

  // ---- bf16 plane region (preferred path) ----
  __bf16* q = (__bf16*)p;
  __bf16* xnh  = q; q += ND;   __bf16* xnl  = q; q += ND;
  __bf16* hh   = q; q += NDF;  __bf16* hl   = q; q += NDF;
  __bf16* awh  = q; q += DD;   __bf16* awl  = q; q += DD;    // abel_w^T planes
  __bf16* w1th = q; q += NEXP * WE;  __bf16* w1tl = q; q += NEXP * WE;
  __bf16* w2th = q; q += NEXP * WE;  __bf16* w2tl = q; q += NEXP * WE;
  __bf16* moeh = q; q += ND;   __bf16* moel = q; q += ND;
  __bf16* r2h  = q; q += DD;   __bf16* r2l  = q; q += DD;
  const size_t need_full = (size_t)((char*)q - (char*)d_ws);
  const bool full = ws_size >= need_full;

  // 1) LayerNorm
  layernorm_kernel<<<NTOK, 256, 0, stream>>>(x, ln_g, ln_b, xn);

  // 2) Gate logits + top-2 softmax -> dense expert weights
  gate_kernel<<<NTOK, 256, 0, stream>>>(xn, gate_w, gate_b, logits);
  top2_kernel<<<(NTOK + 255) / 256, 256, 0, stream>>>(logits, wdense);

  hipMemsetAsync(moe, 0, ND * sizeof(float), stream);

  if (full) {
    // --- preferred path: hi/lo planes + async-LDS GEMMs ---
    launch_decompose(xn, xnh, xnl, (long)ND, stream);
    launch_decompose_t(abel_w, awh, awl, DMODEL, DMODEL, stream);
    for (int e = 0; e < NEXP; ++e) {
      launch_decompose_t(w1 + (size_t)e * WE, w1th + (size_t)e * WE,
                         w1tl + (size_t)e * WE, DMODEL, DFF, stream);
      launch_decompose_t(w2 + (size_t)e * WE, w2th + (size_t)e * WE,
                         w2tl + (size_t)e * WE, DFF, DMODEL, stream);
    }

    // 3) abel = gelu(xn @ abel_w + abel_b)
    launch_gemm_planes<EPI_GELU>(xnh, xnl, awh, awl, abel, nullptr, nullptr,
                                 abel_b, nullptr, 0.f,
                                 NTOK, DMODEL, DMODEL, stream);

    // 4) MoE (h emitted directly as hi/lo planes by the GELU epilogue)
    for (int e = 0; e < NEXP; ++e) {
      launch_gemm_planes<EPI_GELU_SPLIT>(xnh, xnl,
                                         w1th + (size_t)e * WE, w1tl + (size_t)e * WE,
                                         nullptr, hh, hl, b1 + (size_t)e * DFF,
                                         nullptr, 0.f, NTOK, DFF, DMODEL, stream);
      launch_gemm_planes<EPI_MOE>(hh, hl,
                                  w2th + (size_t)e * WE, w2tl + (size_t)e * WE,
                                  moe, nullptr, nullptr, b2 + (size_t)e * DMODEL,
                                  wdense + (size_t)e * NTOK, 0.f,
                                  NTOK, DMODEL, DFF, stream);
    }
  } else {
    // --- fallback path: fused f32->bf16 split inside the GEMM ---
    launch_gemm_bf3<EPI_GELU>(xn, abel_w, abel, abel_b, nullptr, 0.f,
                              NTOK, DMODEL, DMODEL, stream);
    for (int e = 0; e < NEXP; ++e) {
      launch_gemm_bf3<EPI_GELU>(xn, w1 + (size_t)e * WE, h_f32,
                                b1 + (size_t)e * DFF, nullptr, 0.f,
                                NTOK, DFF, DMODEL, stream);
      launch_gemm_bf3<EPI_MOE>(h_f32, w2 + (size_t)e * WE, moe,
                               b2 + (size_t)e * DMODEL,
                               wdense + (size_t)e * NTOK, 0.f,
                               NTOK, DMODEL, DFF, stream);
    }
  }

  // 5) expm(gens[i]) via scaling-and-squaring on the exact f32 WMMA path:
  //    X = G/16; Horner P <- (X@P)/k + I (k=7..1, P0 = X/8 + I); square x4.
  const int nDD = (int)DD;
  for (int i = 0; i < 3; ++i) {
    scale_kernel<<<(nDD + 255) / 256, 256, 0, stream>>>(
        gens + (size_t)i * DD, Xb, 1.0f / 16.0f, nDD);
    scale_ident_kernel<<<(nDD + 255) / 256, 256, 0, stream>>>(
        Xb, Pb, 1.0f / 8.0f, DMODEL);
    float* P = Pb; float* T = Tb;
    for (int k = 7; k >= 1; --k) {
      launch_gemm_f32<EPI_SCALE_IDENT>(Xb, P, T, nullptr, nullptr,
                                       1.0f / (float)k,
                                       DMODEL, DMODEL, DMODEL, stream);
      float* tp = P; P = T; T = tp;
    }
    for (int sq = 0; sq < 4; ++sq) {
      launch_gemm_f32<EPI_NONE>(P, P, T, nullptr, nullptr, 0.f,
                                DMODEL, DMODEL, DMODEL, stream);
      float* tp = P; P = T; T = tp;
    }
    hipMemcpyAsync(Em[i], P, DD * sizeof(float), hipMemcpyDeviceToDevice, stream);
  }

  // 6) rot = E0 @ E1 @ E2 (exact f32); nonabel = moe @ rot (split-bf16)
  launch_gemm_f32<EPI_NONE>(Em[0], Em[1], R1, nullptr, nullptr, 0.f,
                            DMODEL, DMODEL, DMODEL, stream);
  launch_gemm_f32<EPI_NONE>(R1, Em[2], R2, nullptr, nullptr, 0.f,
                            DMODEL, DMODEL, DMODEL, stream);
  if (full) {
    launch_decompose(moe, moeh, moel, (long)ND, stream);
    launch_decompose_t(R2, r2h, r2l, DMODEL, DMODEL, stream);
    launch_gemm_planes<EPI_NONE>(moeh, moel, r2h, r2l, nonabel, nullptr, nullptr,
                                 nullptr, nullptr, 0.f,
                                 NTOK, DMODEL, DMODEL, stream);
  } else {
    launch_gemm_bf3<EPI_NONE>(moe, R2, nonabel, nullptr, nullptr, 0.f,
                              NTOK, DMODEL, DMODEL, stream);
  }

  // 7) Sigmoid-gated mix + residual
  mix_out_kernel<<<NTOK, 256, 0, stream>>>(x, abel, nonabel, mix_w, mix_b, out);
}
